// ClassificationModel_5394478924436
// MI455X (gfx1250) — matrix operand order repaired and checked by reading, not compiled, not run on hardware
//
#include <hip/hip_runtime.h>
#include <math.h>

// ---------------- problem constants ----------------
#define NN     100000
#define EE     1600000LL
#define F_IN   128
#define GFD    7
#define HH     128
#define GG     2000
#define XCOL   135            // F_IN + GF
#define NF     ((long long)NN * 128)

// ---------------- WMMA types ----------------
typedef __attribute__((ext_vector_type(16))) __bf16 v16bf;
typedef __attribute__((ext_vector_type(8)))  float  v8f;
typedef __attribute__((ext_vector_type(4)))  float  f32x4;

union V16 { v16bf v; f32x4 q[2]; };

__device__ __forceinline__ __bf16 f32_to_bf16(float f) {
  union { float f; unsigned u; } a; a.f = f;
  unsigned r = (a.u + 0x7FFFu + ((a.u >> 16) & 1u)) >> 16;  // RNE
  union { unsigned short s; __bf16 b; } o; o.s = (unsigned short)r;
  return o.b;
}

// ---------------- generic fill / convert ----------------
__global__ void fill_f32(float* p, float v, long long n) {
  long long i = (long long)blockIdx.x * blockDim.x + threadIdx.x;
  if (i < n) p[i] = v;
}

__global__ void cvt_f32_bf16(const float* __restrict__ src, __bf16* __restrict__ dst, long long n) {
  long long i = (long long)blockIdx.x * blockDim.x + threadIdx.x;
  if (i < n) dst[i] = f32_to_bf16(src[i]);
}

// pack x_all[:, :128] (stride 135) into bf16 N x 128 (stride 128)
__global__ void xprep_kernel(const float* __restrict__ x_all, __bf16* __restrict__ x_bf) {
  long long t = (long long)blockIdx.x * blockDim.x + threadIdx.x;
  if (t >= NF) return;
  int i = (int)(t >> 7), c = (int)(t & 127);
  x_bf[t] = f32_to_bf16(x_all[(long long)i * XCOL + c]);
}

// transpose weight W[K, ncol] (row-major, ld = ncol) -> Wt[ncol, K] bf16 (col-major per output col)
__global__ void wprep_kernel(const float* __restrict__ W, int ldw, int K, int ncol,
                             __bf16* __restrict__ Wt) {
  int t = blockIdx.x * blockDim.x + threadIdx.x;
  if (t >= K * ncol) return;
  int c = t / K, k = t - c * K;
  Wt[(long long)c * K + k] = f32_to_bf16(W[(long long)k * ldw + c]);
}

// ---------------- degree / norm precompute ----------------
__global__ void dinv_kernel(const float* __restrict__ x_all, float* __restrict__ dinv) {
  int i = blockIdx.x * blockDim.x + threadIdx.x;
  if (i >= NN) return;
  float d = 1.0f / (x_all[(long long)i * XCOL + F_IN] + 1.0f);
  dinv[i] = fminf(d, 1.0f);
}

__global__ void deg_edges_kernel(const int* __restrict__ ei, float* __restrict__ deg) {
  long long e = (long long)blockIdx.x * blockDim.x + threadIdx.x;
  if (e >= EE) return;
  atomicAdd(&deg[ei[EE + e]], 1.0f);
}

__global__ void deg_finish_kernel(float* __restrict__ deg_invdeg, float* __restrict__ dis) {
  int i = blockIdx.x * blockDim.x + threadIdx.x;
  if (i >= NN) return;
  float d = deg_invdeg[i];
  dis[i] = rsqrtf(d);
  deg_invdeg[i] = 1.0f / d;   // becomes inv_deg
}

__global__ void norm_edges_kernel(const int* __restrict__ ei, const float* __restrict__ dis,
                                  float* __restrict__ norme) {
  long long e = (long long)blockIdx.x * blockDim.x + threadIdx.x;
  if (e >= EE) return;
  norme[e] = dis[ei[e]] * dis[ei[EE + e]];
}

// ---------------- WMMA bf16 GEMM ----------------
// out[M,ncol] = A_bf16[M,K](lda) @ Wt_bf16[ncol,K](col-major) + bias, optional relu,
// optional bf16 mirror. One wave: 16 rows x NT*16 cols. Per k-chunk: issue ALL
// fragment loads first (one big clause), then the NT WMMAs back-to-back.
template <int NT>
__global__ __launch_bounds__(128)
void gemm_bf16(const __bf16* __restrict__ A, int lda,
               const __bf16* __restrict__ Wt,
               const float* __restrict__ bias,
               float* __restrict__ out, __bf16* __restrict__ out_bf, int ldo,
               int M, int K, int relu) {
  const int lane  = threadIdx.x & 31;
  const int wv    = threadIdx.x >> 5;
  const int rt    = blockIdx.x * 4 + wv;
  if (rt * 16 >= M) return;                    // wave-uniform exit
  const int row0  = rt * 16;
  const int col0  = blockIdx.y * (NT * 16);
  const int lhalf = lane >> 4;                  // 0 | 1
  const int lmod  = lane & 15;

  v8f acc[NT];
  #pragma unroll
  for (int t = 0; t < NT; ++t)
    #pragma unroll
    for (int d = 0; d < 8; ++d) acc[t][d] = 0.0f;

  // A: lane holds row (row0+lmod); K chunk lhalf*16 .. +16 contiguous
  const __bf16* abase = A + (long long)(row0 + lmod) * lda + (lhalf << 4);
  // B: lane holds output column; K chunk lhalf*16 .. +16 contiguous in col-major Wt
  const __bf16* bbase[NT];
  #pragma unroll
  for (int t = 0; t < NT; ++t)
    bbase[t] = Wt + (long long)(col0 + t * 16 + lmod) * K + (lhalf << 4);

  for (int kc = 0; kc < K; kc += 32) {
    V16 ua;
    V16 ub[NT];
    ua.q[0] = *reinterpret_cast<const f32x4*>(abase + kc);
    ua.q[1] = *reinterpret_cast<const f32x4*>(abase + kc + 8);
    #pragma unroll
    for (int t = 0; t < NT; ++t) {
      ub[t].q[0] = *reinterpret_cast<const f32x4*>(bbase[t] + kc);
      ub[t].q[1] = *reinterpret_cast<const f32x4*>(bbase[t] + kc + 8);
    }
    #pragma unroll
    for (int t = 0; t < NT; ++t)
      acc[t] = __builtin_amdgcn_wmma_f32_16x16x32_bf16(
          false, ua.v, false, ub[t].v, (short)0, acc[t], false, false);
  }

  // D layout: vgpr d, lanes 0-15 -> row row0+d, lanes 16-31 -> row row0+d+8; col = col0+t*16+lmod
  #pragma unroll
  for (int t = 0; t < NT; ++t) {
    int c = col0 + t * 16 + lmod;
    float bv = bias[c];
    #pragma unroll
    for (int d = 0; d < 8; ++d) {
      int r = row0 + d + (lhalf << 3);
      float v = acc[t][d] + bv;
      if (relu) v = fmaxf(v, 0.0f);
      out[(long long)r * ldo + c] = v;
      if (out_bf) out_bf[(long long)r * ldo + c] = f32_to_bf16(v);
    }
  }
}

// ---------------- conv4 hw: raw_v2(N x 7) @ W4[128:135,:] + b4 ----------------
__global__ void conv4_hw_kernel(const float* __restrict__ x_all, const float* __restrict__ W4,
                                const float* __restrict__ b4, float* __restrict__ hwb) {
  long long t = (long long)blockIdx.x * blockDim.x + threadIdx.x;
  if (t >= NF) return;
  int i = (int)(t >> 7), c = (int)(t & 127);
  const float* v = x_all + (long long)i * XCOL + F_IN;
  const float* w = W4 + (long long)F_IN * HH + c;
  float s = b4[c];
  #pragma unroll
  for (int k = 0; k < GFD; ++k) s += v[k] * w[k * HH];
  hwb[t] = s;
}

// ---------------- edge scatter: acc2 += hw[src]; acc1 += hw[src]*norm_e ----------------
__global__ void scatter_kernel(const float* __restrict__ hw, const int* __restrict__ ei,
                               const float* __restrict__ norme,
                               float* __restrict__ acc1, float* __restrict__ acc2) {
  long long t = (long long)blockIdx.x * blockDim.x + threadIdx.x;
  if (t >= EE * 32) return;
  long long e = t >> 5;
  int q = (int)(t & 31);
  int src = ei[e], dst = ei[EE + e];
  float ne = norme[e];
  const f32x4 g = *reinterpret_cast<const f32x4*>(hw + (long long)src * 128 + q * 4);
  long long o = (long long)dst * 128 + q * 4;
  atomicAdd(acc2 + o + 0, g.x); atomicAdd(acc2 + o + 1, g.y);
  atomicAdd(acc2 + o + 2, g.z); atomicAdd(acc2 + o + 3, g.w);
  atomicAdd(acc1 + o + 0, g.x * ne); atomicAdd(acc1 + o + 1, g.y * ne);
  atomicAdd(acc1 + o + 2, g.z * ne); atomicAdd(acc1 + o + 3, g.w * ne);
}

// ---------------- combine: cross-stream mix + self-loop + relu (+ bf16 mirror) ----------------
__global__ void combine_kernel(const float* __restrict__ acc1a, const float* __restrict__ acc2a,
                               const float* __restrict__ acc1b, const float* __restrict__ acc2b,
                               const float* __restrict__ hwa, const float* __restrict__ hwb,
                               const float* __restrict__ invdeg, const float* __restrict__ dinv,
                               float* __restrict__ outa, float* __restrict__ outb,
                               __bf16* __restrict__ outa_bf, __bf16* __restrict__ outb_bf) {
  long long t = (long long)blockIdx.x * blockDim.x + threadIdx.x;
  if (t >= NF) return;
  int i = (int)(t >> 7);
  float id = invdeg[i], dv = dinv[i];
  float va = fmaxf(acc1a[t] + hwa[t] * id + acc2b[t] * dv, 0.0f);
  float vb = fmaxf(acc1b[t] + hwb[t] * id + acc2a[t] * dv, 0.0f);
  outa[t] = va;  outb[t] = vb;
  outa_bf[t] = f32_to_bf16(va);
  outb_bf[t] = f32_to_bf16(vb);
}

// ---------------- per-graph counts ----------------
__global__ void counts_kernel(const int* __restrict__ batch, float* __restrict__ counts) {
  int i = blockIdx.x * blockDim.x + threadIdx.x;
  if (i >= NN) return;
  atomicAdd(&counts[batch[i]], 1.0f);
}

// ---------------- segment-max pooling (relu outputs >= 0 -> uint atomicMax valid) ----------------
__global__ void pool_kernel(const float* __restrict__ x1, const float* __restrict__ z1,
                            const float* __restrict__ v1, const float* __restrict__ x2,
                            const float* __restrict__ z2, const float* __restrict__ v2,
                            const int* __restrict__ batch, float* __restrict__ zfinal) {
  long long t = (long long)blockIdx.x * blockDim.x + threadIdx.x;
  if (t >= NF) return;
  int i = (int)(t >> 7), f = (int)(t & 127);
  int g = batch[i];
  unsigned* zf = reinterpret_cast<unsigned*>(zfinal + (long long)g * 768);
  atomicMax(zf +   0 + f, __float_as_uint(x1[t]));
  atomicMax(zf + 128 + f, __float_as_uint(z1[t]));
  atomicMax(zf + 256 + f, __float_as_uint(v1[t]));
  atomicMax(zf + 384 + f, __float_as_uint(x2[t]));
  atomicMax(zf + 512 + f, __float_as_uint(z2[t]));
  atomicMax(zf + 640 + f, __float_as_uint(v2[t]));
}

// ---------------- fl_l4: fg(G x 6) @ W(6x64) + b, relu ----------------
__global__ void fl4_kernel(const float* __restrict__ graph_y, const float* __restrict__ counts,
                           const float* __restrict__ W, const float* __restrict__ b,
                           float* __restrict__ out, __bf16* __restrict__ out_bf /* +512, ld 640 */) {
  int t = blockIdx.x * blockDim.x + threadIdx.x;
  if (t >= GG * 64) return;
  int g = t >> 6, c = t & 63;
  float fg[6];
  fg[0] = graph_y[(long long)g * GFD + 0] / counts[g];
  #pragma unroll
  for (int k = 1; k < 6; ++k) fg[k] = graph_y[(long long)g * GFD + k];
  float s = b[c];
  #pragma unroll
  for (int k = 0; k < 6; ++k) s += fg[k] * W[k * 64 + c];
  s = fmaxf(s, 0.0f);
  out[(long long)g * 640 + c] = s;
  out_bf[(long long)g * 640 + c] = f32_to_bf16(s);
}

// ---------------- cO_l3: h2(G x 64) @ W(64x2) + b ----------------
__global__ void col3_kernel(const float* __restrict__ h2, const float* __restrict__ W,
                            const float* __restrict__ b, float* __restrict__ logits) {
  int t = blockIdx.x * blockDim.x + threadIdx.x;
  if (t >= GG * 2) return;
  int g = t >> 1, c = t & 1;
  float s = b[c];
  #pragma unroll
  for (int k = 0; k < 64; ++k) s += h2[(long long)g * 64 + k] * W[k * 2 + c];
  logits[t] = s;
}

// ---------------- final: log-softmax, loss, pred, probs, acc ----------------
__global__ void final_kernel(const float* __restrict__ logits, const int* __restrict__ label,
                             float* __restrict__ out) {
  int g = blockIdx.x * blockDim.x + threadIdx.x;
  if (g >= GG) return;
  float l0 = logits[g * 2 + 0], l1 = logits[g * 2 + 1];
  float m = fmaxf(l0, l1);
  float lse = m + logf(expf(l0 - m) + expf(l1 - m));
  float p0 = expf(l0 - lse), p1 = expf(l1 - lse);
  int pred = (l1 > l0) ? 1 : 0;
  out[1 + g] = (float)pred;
  out[1 + GG + g * 2 + 0] = p0;
  out[1 + GG + g * 2 + 1] = p1;
  int lb = label[g];
  float lp = ((lb == 0) ? l0 : l1) - lse;
  atomicAdd(&out[0], -lp / (float)GG);
  if (pred == lb) atomicAdd(&out[1 + 3 * GG], 1.0f);
}

// =====================================================================
static inline void launch_gemm(const __bf16* A, int lda, const __bf16* Wt, const float* bias,
                               float* out, __bf16* out_bf, int ldo,
                               int M, int K, int ncol, int relu, hipStream_t s) {
  unsigned gx = (unsigned)(((M + 15) / 16 + 3) / 4);
  if (ncol % 64 == 0) {
    gemm_bf16<4><<<dim3(gx, (unsigned)(ncol / 64)), 128, 0, s>>>(A, lda, Wt, bias, out, out_bf, ldo, M, K, relu);
  } else {
    gemm_bf16<2><<<dim3(gx, (unsigned)(ncol / 32)), 128, 0, s>>>(A, lda, Wt, bias, out, out_bf, ldo, M, K, relu);
  }
}

static inline void launch_wprep(const float* W, int ldw, int K, int ncol, __bf16* Wt, hipStream_t s) {
  unsigned g = (unsigned)((K * ncol + 255) / 256);
  wprep_kernel<<<g, 256, 0, s>>>(W, ldw, K, ncol, Wt);
}

extern "C" void kernel_launch(void* const* d_in, const int* in_sizes, int n_in,
                              void* d_out, int out_size, void* d_ws, size_t ws_size,
                              hipStream_t stream) {
  const float* x_all   = (const float*)d_in[0];
  const int*   ei      = (const int*)d_in[1];
  const int*   batch   = (const int*)d_in[2];
  const int*   label   = (const int*)d_in[3];
  const float* graph_y = (const float*)d_in[4];

  enum { Pconv1 = 0, Pconv2, Pconv3, Pconv4, Pconv5, Pconv6,
         Pc1_l1, Pc1_l2, Pc1_l3, PcO_l1, PcO_l2, PcO_l3,
         Pfl_l1, Pfl_l2, Pfl_l3, Pfl_l4, Pfl_l5, Pfl_l6 };
  #define PW(i) ((const float*)d_in[5 + 2 * (i)])
  #define PB(i) ((const float*)d_in[6 + 2 * (i)])

  // ---- workspace carve-up: f32 region then bf16 region (all sizes keep 16B alignment) ----
  float* ws = (float*)d_ws;
  long long o = 0;
  float* hw_a  = ws + o; o += NF;
  float* hw_b  = ws + o; o += NF;
  float* acc1a = ws + o; o += NF;
  float* acc2a = ws + o; o += NF;
  float* acc1b = ws + o; o += NF;
  float* acc2b = ws + o; o += NF;
  float* x1 = ws + o; o += NF;
  float* x2 = ws + o; o += NF;
  float* z1 = ws + o; o += NF;
  float* z2 = ws + o; o += NF;
  float* v1 = ws + o; o += NF;
  float* v2 = ws + o; o += NF;
  float* invdeg = ws + o; o += NN;     // deg -> 1/deg
  float* dis    = ws + o; o += NN;
  float* dinv   = ws + o; o += NN;
  float* norme  = ws + o; o += EE;
  float* counts = ws + o; o += GG;
  float* zfinal = ws + o; o += (long long)GG * 768;
  float* fea    = ws + o; o += (long long)GG * 640;
  float* h1     = ws + o; o += (long long)GG * 256;
  float* h2     = ws + o; o += (long long)GG * 64;
  float* logits = ws + o; o += (long long)GG * 2;

  __bf16* bws = (__bf16*)(ws + o);
  long long b = 0;
  __bf16* x_bf   = bws + b; b += NF;
  __bf16* a_bf   = bws + b; b += NF;           // bf16 of current a-stream activation
  __bf16* b_bf   = bws + b; b += NF;           // bf16 of current b-stream activation
  __bf16* zf_bf  = bws + b; b += (long long)GG * 768;
  __bf16* fea_bf = bws + b; b += (long long)GG * 640;
  __bf16* h1_bf  = bws + b; b += (long long)GG * 256;
  __bf16* wt_c1  = bws + b; b += 128 * 128;
  __bf16* wt_c2  = bws + b; b += 128 * 128;
  __bf16* wt_c3  = bws + b; b += 128 * 128;
  __bf16* wt_c5  = bws + b; b += 128 * 128;
  __bf16* wt_c6  = bws + b; b += 128 * 128;
  __bf16* wt_f1  = bws + b; b += 768 * 256;
  __bf16* wt_f2  = bws + b; b += 256 * 128;
  __bf16* wt_f3  = bws + b; b += 128 * 128;
  __bf16* wt_f5  = bws + b; b += 64 * 32;
  __bf16* wt_f6  = bws + b; b += 32 * 32;
  __bf16* wt_o1  = bws + b; b += 640 * 256;
  __bf16* wt_o2  = bws + b; b += 256 * 64;

  float* dout = (float*)d_out;

  const int TB = 256;
  const unsigned gNF  = (unsigned)((NF + TB - 1) / TB);
  const unsigned gE   = (unsigned)((EE + TB - 1) / TB);
  const unsigned gE32 = (unsigned)((EE * 32 + TB - 1) / TB);
  const unsigned gN   = (unsigned)((NN + TB - 1) / TB);

  // ---- weight prep (bf16 + transpose; tiny, runs every call for determinism) ----
  launch_wprep(PW(Pconv1), HH, 128, HH, wt_c1, stream);   // rows 0..127 of (135,128)
  launch_wprep(PW(Pconv2), HH, 128, HH, wt_c2, stream);
  launch_wprep(PW(Pconv3), HH, 128, HH, wt_c3, stream);
  launch_wprep(PW(Pconv5), HH, 128, HH, wt_c5, stream);
  launch_wprep(PW(Pconv6), HH, 128, HH, wt_c6, stream);
  launch_wprep(PW(Pfl_l1), 256, 768, 256, wt_f1, stream);
  launch_wprep(PW(Pfl_l2), 128, 256, 128, wt_f2, stream);
  launch_wprep(PW(Pfl_l3), 128, 128, 128, wt_f3, stream);
  launch_wprep(PW(Pfl_l5), 32, 64, 32, wt_f5, stream);
  launch_wprep(PW(Pfl_l6), 32, 32, 32, wt_f6, stream);
  launch_wprep(PW(PcO_l1), 256, 640, 256, wt_o1, stream);
  launch_wprep(PW(PcO_l2), 64, 256, 64, wt_o2, stream);

  // ---- degrees / norms / D_inv / x feature pack ----
  dinv_kernel<<<gN, TB, 0, stream>>>(x_all, dinv);
  fill_f32<<<gN, TB, 0, stream>>>(invdeg, 1.0f, NN);           // deg starts at 1 (self loop)
  deg_edges_kernel<<<gE, TB, 0, stream>>>(ei, invdeg);
  deg_finish_kernel<<<gN, TB, 0, stream>>>(invdeg, dis);
  norm_edges_kernel<<<gE, TB, 0, stream>>>(ei, dis, norme);
  xprep_kernel<<<gNF, TB, 0, stream>>>(x_all, x_bf);

  // ---- 3 conv stages ----
  const __bf16* wta[3] = { wt_c1, wt_c2, wt_c3 };
  const __bf16* wtb[3] = { nullptr, wt_c5, wt_c6 };
  const int bA[3] = { Pconv1, Pconv2, Pconv3 };
  const int bB[3] = { Pconv4, Pconv5, Pconv6 };
  float* out_a[3] = { x1, z1, v1 };
  float* out_b[3] = { x2, z2, v2 };

  for (int s = 0; s < 3; ++s) {
    const __bf16* ina = (s == 0) ? x_bf : a_bf;
    const __bf16* inb = b_bf;
    launch_gemm(ina, 128, wta[s], PB(bA[s]), hw_a, nullptr, HH, NN, 128, HH, 0, stream);
    if (s == 0)
      conv4_hw_kernel<<<gNF, TB, 0, stream>>>(x_all, PW(Pconv4), PB(Pconv4), hw_b);
    else
      launch_gemm(inb, 128, wtb[s], PB(bB[s]), hw_b, nullptr, HH, NN, 128, HH, 0, stream);

    fill_f32<<<gNF, TB, 0, stream>>>(acc1a, 0.0f, NF);
    fill_f32<<<gNF, TB, 0, stream>>>(acc2a, 0.0f, NF);
    fill_f32<<<gNF, TB, 0, stream>>>(acc1b, 0.0f, NF);
    fill_f32<<<gNF, TB, 0, stream>>>(acc2b, 0.0f, NF);
    scatter_kernel<<<gE32, TB, 0, stream>>>(hw_a, ei, norme, acc1a, acc2a);
    scatter_kernel<<<gE32, TB, 0, stream>>>(hw_b, ei, norme, acc1b, acc2b);
    combine_kernel<<<gNF, TB, 0, stream>>>(acc1a, acc2a, acc1b, acc2b, hw_a, hw_b,
                                           invdeg, dinv, out_a[s], out_b[s], a_bf, b_bf);
  }

  // ---- pooling ----
  fill_f32<<<(unsigned)(((long long)GG * 768 + TB - 1) / TB), TB, 0, stream>>>(zfinal, 0.0f, (long long)GG * 768);
  fill_f32<<<(unsigned)((GG + TB - 1) / TB), TB, 0, stream>>>(counts, 0.0f, GG);
  counts_kernel<<<gN, TB, 0, stream>>>(batch, counts);
  pool_kernel<<<gNF, TB, 0, stream>>>(x1, z1, v1, x2, z2, v2, batch, zfinal);
  cvt_f32_bf16<<<(unsigned)(((long long)GG * 768 + TB - 1) / TB), TB, 0, stream>>>(zfinal, zf_bf, (long long)GG * 768);

  // ---- head: fea layout x1f[0:256) x2f[256:384) x3f[384:512) z1f[512:576) z2f[576:608) z3f[608:640) ----
  launch_gemm(zf_bf, 768, wt_f1, PB(Pfl_l1), fea + 0,   fea_bf + 0,   640, GG, 768, 256, 1, stream);
  launch_gemm(fea_bf + 0,   640, wt_f2, PB(Pfl_l2), fea + 256, fea_bf + 256, 640, GG, 256, 128, 1, stream);
  launch_gemm(fea_bf + 256, 640, wt_f3, PB(Pfl_l3), fea + 384, fea_bf + 384, 640, GG, 128, 128, 1, stream);
  fl4_kernel<<<(unsigned)((GG * 64 + TB - 1) / TB), TB, 0, stream>>>(graph_y, counts, PW(Pfl_l4), PB(Pfl_l4),
                                                                     fea + 512, fea_bf + 512);
  launch_gemm(fea_bf + 512, 640, wt_f5, PB(Pfl_l5), fea + 576, fea_bf + 576, 640, GG, 64, 32, 1, stream);
  launch_gemm(fea_bf + 576, 640, wt_f6, PB(Pfl_l6), fea + 608, fea_bf + 608, 640, GG, 32, 32, 1, stream);

  launch_gemm(fea_bf, 640, wt_o1, PB(PcO_l1), h1, h1_bf, 256, GG, 640, 256, 1, stream);
  launch_gemm(h1_bf, 256, wt_o2, PB(PcO_l2), h2, nullptr, 64, GG, 256, 64, 1, stream);
  col3_kernel<<<(unsigned)((GG * 2 + TB - 1) / TB), TB, 0, stream>>>(h2, PW(PcO_l3), PB(PcO_l3), logits);

  // ---- outputs: [loss(1), pred(G), probs(G*2), acc(1)] ----
  fill_f32<<<1, 1, 0, stream>>>(dout, 0.0f, 1);
  fill_f32<<<1, 1, 0, stream>>>(dout + 1 + 3 * GG, 0.0f, 1);
  final_kernel<<<(unsigned)((GG + TB - 1) / TB), TB, 0, stream>>>(logits, label, dout);
}
